// Model_33921651704541
// MI455X (gfx1250) — matrix operand order repaired and checked by reading, not compiled, not run on hardware
//
#include <hip/hip_runtime.h>

// DenseNet-121 forward for MI455X (gfx1250), wave32 + WMMA bf16 implicit GEMM.
// NHWC activation layout: WMMA K-chunks are contiguous channels -> vectorized
// b128 staging, no per-element im2col decode. BN+ReLU fused into A staging.
// B (weight) tiles staged with CDNA5 async global->LDS (ASYNCcnt) copies.
// Dense-block concat = channel offset in the NHWC innermost dim.

typedef __attribute__((ext_vector_type(16))) __bf16 v16bf;
typedef __attribute__((ext_vector_type(8)))  __bf16 v8bf;
typedef __attribute__((ext_vector_type(8)))  float  v8f;

#define DN_EPS 1e-5f

// ---------------- device helpers ----------------

__device__ __forceinline__ v16bf mk16(v8bf lo, v8bf hi) {
  return __builtin_shufflevector(lo, hi,
      0, 1, 2, 3, 4, 5, 6, 7, 8, 9, 10, 11, 12, 13, 14, 15);
}

// BN+ReLU 8 contiguous channels -> bf16
__device__ __forceinline__ v8bf bnrelu8(const float* __restrict__ src,
                                        const float* __restrict__ sc,
                                        const float* __restrict__ sh) {
  const float4 f0 = *(const float4*)(src);
  const float4 f1 = *(const float4*)(src + 4);
  const float4 s0 = *(const float4*)(sc);
  const float4 s1 = *(const float4*)(sc + 4);
  const float4 h0 = *(const float4*)(sh);
  const float4 h1 = *(const float4*)(sh + 4);
  v8bf r;
  r[0] = (__bf16)fmaxf(f0.x * s0.x + h0.x, 0.0f);
  r[1] = (__bf16)fmaxf(f0.y * s0.y + h0.y, 0.0f);
  r[2] = (__bf16)fmaxf(f0.z * s0.z + h0.z, 0.0f);
  r[3] = (__bf16)fmaxf(f0.w * s0.w + h0.w, 0.0f);
  r[4] = (__bf16)fmaxf(f1.x * s1.x + h1.x, 0.0f);
  r[5] = (__bf16)fmaxf(f1.y * s1.y + h1.y, 0.0f);
  r[6] = (__bf16)fmaxf(f1.z * s1.z + h1.z, 0.0f);
  r[7] = (__bf16)fmaxf(f1.w * s1.w + h1.w, 0.0f);
  return r;
}

// async 16-byte global->LDS copy (per active lane); tracked on ASYNCcnt
__device__ __forceinline__ void async_b128_to_lds(unsigned lds_byte_addr,
                                                  const void* gaddr) {
  asm volatile("global_load_async_to_lds_b128 %0, %1, off"
               :: "v"(lds_byte_addr), "v"(gaddr) : "memory");
}
__device__ __forceinline__ void wait_async0() {
  asm volatile("s_wait_asynccnt 0x0" ::: "memory");
}

// ---------------- small utility kernels ----------------

__global__ __launch_bounds__(256)
void k_f32_to_bf16(const float* __restrict__ s, __bf16* __restrict__ d, int n) {
  int i = blockIdx.x * 256 + threadIdx.x;
  if (i < n) d[i] = (__bf16)s[i];
}

// dense-layer weights [32][cin][3][3] -> bf16 tap-major [o][tap][c]
__global__ __launch_bounds__(256)
void k_wreorder3x3(const float* __restrict__ w, __bf16* __restrict__ d, int cin) {
  int i = blockIdx.x * 256 + threadIdx.x;
  int total = 32 * 9 * cin;
  if (i >= total) return;
  int c = i % cin; int t = i / cin;
  int tap = t % 9; int o = t / 9;
  d[i] = (__bf16)w[(o * cin + c) * 9 + tap];
}

__global__ __launch_bounds__(256)
void k_bn_fold(const float* __restrict__ g, const float* __restrict__ b,
               const float* __restrict__ m, const float* __restrict__ v,
               float* __restrict__ scale, float* __restrict__ shift, int c) {
  int i = blockIdx.x * 256 + threadIdx.x;
  if (i < c) {
    float s = g[i] * rsqrtf(v[i] + DN_EPS);
    scale[i] = s;
    shift[i] = b[i] - m[i] * s;
  }
}

// stem: BN+ReLU then 3x3/s2/p1 maxpool; NCHW in -> NHWC out (channel offset 0)
__global__ __launch_bounds__(256)
void k_bnrelu_maxpool_nhwc(const float* __restrict__ in, const float* __restrict__ scale,
                           const float* __restrict__ shift, float* __restrict__ out,
                           int N, int C, int H, int W, int Ho, int Wo, int CoutTot) {
  int idx = blockIdx.x * 256 + threadIdx.x;
  int total = N * Ho * Wo * C;
  if (idx >= total) return;
  int c  = idx % C;  int t = idx / C;
  int xo = t % Wo;   t /= Wo;
  int yo = t % Ho;   int n = t / Ho;
  float s = scale[c], sh = shift[c];
  float mx = -3.4e38f;
  for (int ky = 0; ky < 3; ++ky) {
    int iy = yo * 2 - 1 + ky;
    if ((unsigned)iy >= (unsigned)H) continue;
    for (int kx = 0; kx < 3; ++kx) {
      int ix = xo * 2 - 1 + kx;
      if ((unsigned)ix >= (unsigned)W) continue;
      float v = in[(((long)n * C + c) * H + iy) * W + ix] * s + sh;
      mx = fmaxf(mx, v);
    }
  }
  mx = fmaxf(mx, 0.0f);
  out[(((long)n * Ho + yo) * Wo + xo) * CoutTot + c] = mx;
}

// NHWC 2x2 average pool; writes channel range [0,C) with stride CoutTot
__global__ __launch_bounds__(256)
void k_avgpool2_nhwc(const float* __restrict__ in, float* __restrict__ out,
                     int N, int C, int H, int W, int CoutTot) {
  int Ho = H >> 1, Wo = W >> 1;
  int idx = blockIdx.x * 256 + threadIdx.x;
  int total = N * Ho * Wo * C;
  if (idx >= total) return;
  int c  = idx % C;  int t = idx / C;
  int xo = t % Wo;   t /= Wo;
  int yo = t % Ho;   int n = t / Ho;
  const float* p = in + ((long)(n * H + yo * 2) * W + xo * 2) * C + c;
  float s = 0.25f * (p[0] + p[C] + p[(long)W * C] + p[(long)W * C + C]);
  out[(((long)n * Ho + yo) * Wo + xo) * CoutTot + c] = s;
}

// final BN+ReLU + global average pool over NHWC input
__global__ __launch_bounds__(256)
void k_bnrelu_gap_nhwc(const float* __restrict__ in, const float* __restrict__ scale,
                       const float* __restrict__ shift, float* __restrict__ out,
                       int N, int C, int H, int W, int CinTot) {
  int idx = blockIdx.x * 256 + threadIdx.x;
  if (idx >= N * C) return;
  int c = idx % C, n = idx / C;
  float s = scale[c], sh = shift[c], acc = 0.0f;
  const float* p = in + (long)n * H * W * CinTot + c;
  for (int i = 0; i < H * W; ++i) acc += fmaxf(p[(long)i * CinTot] * s + sh, 0.0f);
  out[n * C + c] = acc / (float)(H * W);
}

__global__ __launch_bounds__(256)
void k_fc(const float* __restrict__ pooled, const float* __restrict__ w,
          const float* __restrict__ b, float* __restrict__ out,
          int N, int C, int O) {
  int idx = blockIdx.x * 256 + threadIdx.x;
  if (idx >= N * O) return;
  int o = idx % O, n = idx / O;
  float acc = b[o];
  const float* pv = pooled + n * C;
  const float* pw = w + o * C;
  for (int c = 0; c < C; ++c) acc += pv[c] * pw[c];
  out[n * O + o] = acc;
}

// ------------- generic implicit-GEMM conv (stem only; NCHW in/out) -------------
__global__ __launch_bounds__(256)
void k_conv_wmma(const float* __restrict__ in, const __bf16* __restrict__ wt,
                 float* __restrict__ out,
                 int N, int Cin, int H, int W,
                 int Cout, int Ho, int Wo,
                 int KH, int KW, int stride, int pad) {
  const int M = N * Ho * Wo;
  const int K = Cin * KH * KW;
  const int KHW = KH * KW;
  const int tilesN = (Cout + 31) >> 5;
  const int tileN = blockIdx.x % tilesN;
  const int tileM = blockIdx.x / tilesN;

  __shared__ __bf16 As[64][40];
  __shared__ __bf16 Bs[32][40];

  const int tid  = threadIdx.x;
  const int lane = tid & 31;
  const int wave = tid >> 5;
  const int wm   = wave >> 1;
  const int wn   = wave & 1;
  const int half = lane >> 4;
  const int l16  = lane & 15;

  const int arow = tid >> 2;
  const int akk0 = (tid & 3) * 8;
  int am = tileM * 64 + arow;
  int an = 0, ayo = 0, axo = 0;
  bool amv = (am < M);
  if (amv) {
    an = am / (Ho * Wo);
    int r = am - an * Ho * Wo;
    ayo = r / Wo;
    axo = r - ayo * Wo;
  }
  const int bkk = tid >> 3;
  const int bj0 = (tid & 7) * 4;

  v8f acc = {};
  for (int kc = 0; kc < K; kc += 32) {
#pragma unroll
    for (int j = 0; j < 8; ++j) {
      int k = kc + akk0 + j;
      float v = 0.0f;
      if (amv && k < K) {
        int c  = k / KHW;
        int r9 = k - c * KHW;
        int ky = r9 / KW;
        int kx = r9 - ky * KW;
        int iy = ayo * stride - pad + ky;
        int ix = axo * stride - pad + kx;
        if ((unsigned)iy < (unsigned)H && (unsigned)ix < (unsigned)W)
          v = in[(((long)an * Cin + c) * H + iy) * W + ix];
      }
      As[arow][akk0 + j] = (__bf16)v;
    }
#pragma unroll
    for (int j = 0; j < 4; ++j) {
      int k   = kc + bkk;
      int col = tileN * 32 + bj0 + j;
      __bf16 v = (__bf16)0.0f;
      if (k < K && col < Cout) v = wt[(long)col * K + k];
      Bs[bkk][bj0 + j] = v;
    }
    __syncthreads();
    v16bf a, b;
#pragma unroll
    for (int e = 0; e < 16; ++e) {
      int ka = half * 8 + (e >> 3) * 16 + (e & 7);
      a[e] = As[wm * 16 + l16][ka];
      int kb = half * 8 + (e >> 3) * 16 + (e & 7);
      b[e] = Bs[kb][wn * 16 + l16];
    }
    acc = __builtin_amdgcn_wmma_f32_16x16x32_bf16(
        false, a, false, b, (short)0, acc, false, false);
    __syncthreads();
  }
  int col = tileN * 32 + wn * 16 + l16;
#pragma unroll
  for (int r = 0; r < 8; ++r) {
    int m = tileM * 64 + wm * 16 + half * 8 + r;
    if (m < M && col < Cout) {
      int n  = m / (Ho * Wo);
      int rr = m - n * Ho * Wo;
      int yo = rr / Wo;
      int xo = rr - yo * Wo;
      out[(((long)n * Cout + col) * Ho + yo) * Wo + xo] = acc[r];
    }
  }
}

// -------- dense-layer 3x3/p1 conv, NHWC, fused BN+ReLU, Cout=32 (WMMA) --------
// K-chunks of 64 channels (two WMMAs per barrier) with a 32-wide tail.
// B tiles copied via async global->LDS.
__global__ __launch_bounds__(256)
void k_conv3x3_nhwc(const float* __restrict__ in,
                    const float* __restrict__ scale, const float* __restrict__ shift,
                    const __bf16* __restrict__ wt,   // [32][9*Cin] tap-major
                    float* __restrict__ out,
                    int N, int Cin, int H, int W, int CinTot, int CoutTot, int co0) {
  const int M = N * H * W;
  const int tid  = threadIdx.x;
  const int lane = tid & 31, wave = tid >> 5;
  const int wm = wave >> 1, wn = wave & 1;
  const int half = lane >> 4, l16 = lane & 15;

  __shared__ alignas(16) __bf16 As[64][72];   // [m-row][k], 144B stride (conflict-free)
  __shared__ alignas(16) __bf16 Bs[32][72];   // [col][k]   (B transposed)

  const int arow = tid >> 2;         // 0..63
  const int ac8  = (tid & 3) * 8;    // 0,8,16,24
  const int am = blockIdx.x * 64 + arow;
  int an = 0, ay = 0, ax = 0;
  const bool amv = (am < M);
  if (amv) { an = am / (H * W); int r = am - an * (H * W); ay = r / W; ax = r - ay * W; }

  const int bcol = tid >> 2;         // tid<128: 0..31
  const int bk0  = (tid & 3) * 8;
  const long K9  = 9L * Cin;
  const unsigned ldsB0 = (unsigned)(size_t)&Bs[bcol & 31][bk0];

  const __bf16* arp = &As[wm * 16 + l16][0];
  const __bf16* brp = &Bs[wn * 16 + l16][0];

  v8f acc = {};
  for (int tap = 0; tap < 9; ++tap) {
    const int ky = tap / 3, kx = tap - (tap / 3) * 3;
    const int iy = ay - 1 + ky, ix = ax - 1 + kx;
    const bool inb = amv && (unsigned)iy < (unsigned)H && (unsigned)ix < (unsigned)W;
    const float* src = in + ((long)(an * H + iy) * W + ix) * CinTot;
    const __bf16* wp0 = wt + (long)bcol * K9 + (long)tap * Cin + bk0;

    int c0 = 0;
    for (; c0 + 64 <= Cin; c0 += 64) {
      v8bf a0 = {}, a1 = {};
      if (inb) {
        a0 = bnrelu8(src + c0 + ac8, scale + c0 + ac8, shift + c0 + ac8);
        a1 = bnrelu8(src + c0 + 32 + ac8, scale + c0 + 32 + ac8, shift + c0 + 32 + ac8);
      }
      *(v8bf*)&As[arow][ac8]      = a0;
      *(v8bf*)&As[arow][32 + ac8] = a1;
      if (tid < 128) {
        async_b128_to_lds(ldsB0,      (const void*)(wp0 + c0));
        async_b128_to_lds(ldsB0 + 64, (const void*)(wp0 + c0 + 32));  // +32 bf16 = 64B
      }
      wait_async0();
      __syncthreads();
#pragma unroll
      for (int kb = 0; kb < 64; kb += 32) {
        v16bf a = mk16(*(const v8bf*)(arp + kb + half * 8),
                       *(const v8bf*)(arp + kb + half * 8 + 16));
        v16bf b = mk16(*(const v8bf*)(brp + kb + half * 8),
                       *(const v8bf*)(brp + kb + half * 8 + 16));
        acc = __builtin_amdgcn_wmma_f32_16x16x32_bf16(
            false, a, false, b, (short)0, acc, false, false);
      }
      __syncthreads();
    }
    if (c0 < Cin) {  // 32-channel tail
      v8bf a0 = {};
      if (inb) a0 = bnrelu8(src + c0 + ac8, scale + c0 + ac8, shift + c0 + ac8);
      *(v8bf*)&As[arow][ac8] = a0;
      if (tid < 128) async_b128_to_lds(ldsB0, (const void*)(wp0 + c0));
      wait_async0();
      __syncthreads();
      v16bf a = mk16(*(const v8bf*)(arp + half * 8),
                     *(const v8bf*)(arp + half * 8 + 16));
      v16bf b = mk16(*(const v8bf*)(brp + half * 8),
                     *(const v8bf*)(brp + half * 8 + 16));
      acc = __builtin_amdgcn_wmma_f32_16x16x32_bf16(
          false, a, false, b, (short)0, acc, false, false);
      __syncthreads();
    }
  }

  const int col = co0 + wn * 16 + l16;
#pragma unroll
  for (int r = 0; r < 8; ++r) {
    int m = blockIdx.x * 64 + wm * 16 + half * 8 + r;
    if (m < M) out[(long)m * CoutTot + col] = acc[r];
  }
}

// -------- transition 1x1 conv, NHWC, fused BN+ReLU (pure GEMM, WMMA) --------
// Cin always a multiple of 64 here: K-chunks of 64, two WMMAs per barrier.
__global__ __launch_bounds__(256)
void k_conv1x1_nhwc(const float* __restrict__ in,
                    const float* __restrict__ scale, const float* __restrict__ shift,
                    const __bf16* __restrict__ wt,   // [Cout][Cin]
                    float* __restrict__ out,
                    int M, int Cin, int Cout, int CoutTot) {
  const int tilesN = Cout >> 5;
  const int tileN = blockIdx.x % tilesN;
  const int tileM = blockIdx.x / tilesN;

  const int tid  = threadIdx.x;
  const int lane = tid & 31, wave = tid >> 5;
  const int wm = wave >> 1, wn = wave & 1;
  const int half = lane >> 4, l16 = lane & 15;

  __shared__ alignas(16) __bf16 As[64][72];
  __shared__ alignas(16) __bf16 Bs[32][72];

  const int arow = tid >> 2;
  const int ac8  = (tid & 3) * 8;
  const int am = tileM * 64 + arow;
  const bool amv = (am < M);
  const float* src = in + (long)(amv ? am : 0) * Cin;

  const int bcol = tid >> 2;
  const int bk0  = (tid & 3) * 8;
  const __bf16* wp0 = wt + (long)(tileN * 32 + bcol) * Cin + bk0;
  const unsigned ldsB0 = (unsigned)(size_t)&Bs[bcol & 31][bk0];

  const __bf16* arp = &As[wm * 16 + l16][0];
  const __bf16* brp = &Bs[wn * 16 + l16][0];

  v8f acc = {};
  for (int c0 = 0; c0 < Cin; c0 += 64) {
    v8bf a0 = {}, a1 = {};
    if (amv) {
      a0 = bnrelu8(src + c0 + ac8, scale + c0 + ac8, shift + c0 + ac8);
      a1 = bnrelu8(src + c0 + 32 + ac8, scale + c0 + 32 + ac8, shift + c0 + 32 + ac8);
    }
    *(v8bf*)&As[arow][ac8]      = a0;
    *(v8bf*)&As[arow][32 + ac8] = a1;
    if (tid < 128) {
      async_b128_to_lds(ldsB0,      (const void*)(wp0 + c0));
      async_b128_to_lds(ldsB0 + 64, (const void*)(wp0 + c0 + 32));
    }
    wait_async0();
    __syncthreads();
#pragma unroll
    for (int kb = 0; kb < 64; kb += 32) {
      v16bf a = mk16(*(const v8bf*)(arp + kb + half * 8),
                     *(const v8bf*)(arp + kb + half * 8 + 16));
      v16bf b = mk16(*(const v8bf*)(brp + kb + half * 8),
                     *(const v8bf*)(brp + kb + half * 8 + 16));
      acc = __builtin_amdgcn_wmma_f32_16x16x32_bf16(
          false, a, false, b, (short)0, acc, false, false);
    }
    __syncthreads();
  }

  const int col = tileN * 32 + wn * 16 + l16;
#pragma unroll
  for (int r = 0; r < 8; ++r) {
    int m = tileM * 64 + wm * 16 + half * 8 + r;
    if (m < M) out[(long)m * CoutTot + col] = acc[r];
  }
}

// ---------------- host orchestration ----------------

extern "C" void kernel_launch(void* const* d_in, const int* in_sizes, int n_in,
                              void* d_out, int out_size, void* d_ws, size_t ws_size,
                              hipStream_t stream) {
  (void)in_sizes; (void)n_in; (void)out_size;
  static const int cfg[4] = {6, 12, 24, 16};
  static const int hb[4]  = {56, 28, 14, 7};
  int cins[58], ctot[4];
  {
    int c = 64, li = 0;
    for (int bi = 0; bi < 4; ++bi) {
      for (int i = 0; i < cfg[bi]; ++i) cins[li++] = c + i * 32;
      c += cfg[bi] * 32;
      ctot[bi] = c;          // 256, 512, 1024, 1024
      if (bi < 3) c /= 2;
    }
  }

  // ---- parse inputs in setup_inputs() insertion order ----
  int ii = 0;
  const float* x = (const float*)d_in[ii++];
  const float* stem_w = (const float*)d_in[ii++];
  const float* stem_bn[4];
  for (int j = 0; j < 4; ++j) stem_bn[j] = (const float*)d_in[ii++];
  const float *lg[58], *lb[58], *lm[58], *lv[58], *lw[58];
  for (int l = 0; l < 58; ++l) {
    lg[l] = (const float*)d_in[ii++]; lb[l] = (const float*)d_in[ii++];
    lm[l] = (const float*)d_in[ii++]; lv[l] = (const float*)d_in[ii++];
    lw[l] = (const float*)d_in[ii++];
  }
  const float *tg[3], *tb[3], *tm[3], *tv[3], *tw[3];
  for (int t = 0; t < 3; ++t) {
    tg[t] = (const float*)d_in[ii++]; tb[t] = (const float*)d_in[ii++];
    tm[t] = (const float*)d_in[ii++]; tv[t] = (const float*)d_in[ii++];
    tw[t] = (const float*)d_in[ii++];
  }
  const float* fin_bn[4];
  for (int j = 0; j < 4; ++j) fin_bn[j] = (const float*)d_in[ii++];
  const float* fcw = (const float*)d_in[ii++];
  const float* fcb = (const float*)d_in[ii++];

  // ---- carve workspace ----
  size_t off = 0;
  char* base = (char*)d_ws;
  auto carve = [&](size_t bytes) -> char* {
    char* p = base + off;
    off += (bytes + 255) & ~(size_t)255;
    return p;
  };
  __bf16* w_stem = (__bf16*)carve((size_t)64 * 147 * sizeof(__bf16));
  __bf16* wlb[58];
  for (int l = 0; l < 58; ++l) wlb[l] = (__bf16*)carve((size_t)32 * cins[l] * 9 * sizeof(__bf16));
  __bf16* wtb[3];
  for (int t = 0; t < 3; ++t) wtb[t] = (__bf16*)carve((size_t)(ctot[t] / 2) * ctot[t] * sizeof(__bf16));
  float* ss_stem = (float*)carve((size_t)64 * 2 * sizeof(float));
  float* ssl[58];
  for (int l = 0; l < 58; ++l) ssl[l] = (float*)carve((size_t)cins[l] * 2 * sizeof(float));
  float* sst[3];
  for (int t = 0; t < 3; ++t) sst[t] = (float*)carve((size_t)ctot[t] * 2 * sizeof(float));
  float* ss_fin = (float*)carve((size_t)1024 * 2 * sizeof(float));
  float* bufA = (float*)carve((size_t)16 * 64 * 112 * 112 * sizeof(float));   // stem NCHW / transition NHWC temp
  float* bufB = (float*)carve((size_t)16 * 56 * 56 * 256 * sizeof(float));    // NHWC concat h buffer
  float* pooled = (float*)carve((size_t)16 * 1024 * sizeof(float));
  if (off > ws_size) return;  // workspace insufficient; nothing safe to do

  auto cdiv = [](long a, long b) { return (int)((a + b - 1) / b); };

  // ---- weight prep ----
  k_f32_to_bf16<<<cdiv(64 * 147, 256), 256, 0, stream>>>(stem_w, w_stem, 64 * 147);
  for (int l = 0; l < 58; ++l) {
    int n = 32 * 9 * cins[l];
    k_wreorder3x3<<<cdiv(n, 256), 256, 0, stream>>>(lw[l], wlb[l], cins[l]);
  }
  for (int t = 0; t < 3; ++t) {
    int n = (ctot[t] / 2) * ctot[t];
    k_f32_to_bf16<<<cdiv(n, 256), 256, 0, stream>>>(tw[t], wtb[t], n);
  }

  // ---- fold BN params into per-channel scale/shift ----
  k_bn_fold<<<1, 64, 0, stream>>>(stem_bn[0], stem_bn[1], stem_bn[2], stem_bn[3],
                                  ss_stem, ss_stem + 64, 64);
  for (int l = 0; l < 58; ++l)
    k_bn_fold<<<cdiv(cins[l], 256), 256, 0, stream>>>(lg[l], lb[l], lm[l], lv[l],
                                                      ssl[l], ssl[l] + cins[l], cins[l]);
  for (int t = 0; t < 3; ++t)
    k_bn_fold<<<cdiv(ctot[t], 256), 256, 0, stream>>>(tg[t], tb[t], tm[t], tv[t],
                                                      sst[t], sst[t] + ctot[t], ctot[t]);
  k_bn_fold<<<cdiv(1024, 256), 256, 0, stream>>>(fin_bn[0], fin_bn[1], fin_bn[2], fin_bn[3],
                                                 ss_fin, ss_fin + 1024, 1024);

  // ---- stem: conv7x7/s2/p3 (NCHW) -> bufA [16,64,112,112] ----
  {
    int M = 16 * 112 * 112, tilesN = 2;
    k_conv_wmma<<<dim3(tilesN * cdiv(M, 64)), 256, 0, stream>>>(
        x, w_stem, bufA, 16, 3, 224, 224, 64, 112, 112, 7, 7, 2, 3);
  }
  // stem BN+ReLU + maxpool -> bufB NHWC [16,56,56,ctot0] channels [0,64)
  {
    long tot = 16L * 64 * 56 * 56;
    k_bnrelu_maxpool_nhwc<<<cdiv(tot, 256), 256, 0, stream>>>(
        bufA, ss_stem, ss_stem + 64, bufB, 16, 64, 112, 112, 56, 56, ctot[0]);
  }

  // ---- dense blocks (NHWC, in-place concat) ----
  int li = 0;
  for (int bi = 0; bi < 4; ++bi) {
    int H = hb[bi], C = ctot[bi];
    for (int i = 0; i < cfg[bi]; ++i, ++li) {
      int cin = cins[li];
      int M = 16 * H * H;
      // reads channels [0,cin), writes channels [cin,cin+32)
      k_conv3x3_nhwc<<<dim3(cdiv(M, 64)), 256, 0, stream>>>(
          bufB, ssl[li], ssl[li] + cin, wlb[li], bufB,
          16, cin, H, H, C, C, cin);
    }
    if (bi < 3) {
      int cout = C / 2;
      int M = 16 * H * H, tilesN = cout / 32;
      // transition: BN+ReLU -> 1x1 conv -> bufA (NHWC, dense channels)
      k_conv1x1_nhwc<<<dim3(tilesN * cdiv(M, 64)), 256, 0, stream>>>(
          bufB, sst[bi], sst[bi] + C, wtb[bi], bufA, M, C, cout, cout);
      // 2x2 avgpool -> bufB channels [0,cout) with next block's channel stride
      long tot = 16L * cout * (H / 2) * (H / 2);
      k_avgpool2_nhwc<<<cdiv(tot, 256), 256, 0, stream>>>(
          bufA, bufB, 16, cout, H, H, ctot[bi + 1]);
    }
  }

  // ---- final BN+ReLU + global avgpool + FC ----
  k_bnrelu_gap_nhwc<<<cdiv(16 * 1024, 256), 256, 0, stream>>>(
      bufB, ss_fin, ss_fin + 1024, pooled, 16, 1024, 7, 7, 1024);
  k_fc<<<1, 256, 0, stream>>>(pooled, fcw, fcb, (float*)d_out, 16, 1024, 10);
}
